// cIGNR_4466765988156
// MI455X (gfx1250) — matrix operand order repaired and checked by reading, not compiled, not run on hardware
//
#include <hip/hip_runtime.h>
#include <hip/hip_bf16.h>
#include <math.h>

// ---------------- static problem config (matches reference) ----------------
#define BATCH   16
#define NBG     128          // nodes per graph / grid side
#define NN_     2048         // total nodes
#define E_      32768        // total edges
#define LATENT  128
#define NPROJ_  50
#define NC_     6573         // NOUT*3
#define NCP_    6576         // padded to 16
#define GRIDPTS 16384        // NBG*NBG

typedef _Float16 half8 __attribute__((ext_vector_type(8)));
typedef _Float16 v16h  __attribute__((ext_vector_type(16)));
typedef float    v8f   __attribute__((ext_vector_type(8)));
typedef unsigned int u32x4 __attribute__((ext_vector_type(4)));
typedef int      i32x4 __attribute__((ext_vector_type(4)));
typedef int      i32x8 __attribute__((ext_vector_type(8)));

union H16u { v16h v; half8 h[2]; };

// ---- WMMA fragment loaders (CDNA5 16-bit layouts, wave32) ----
// A (16x32 f16): lane<16 -> row=lane, K = kb..kb+7 and kb+16..kb+23 with kb=0;
//                lane>=16 -> same row, kb=8.
__device__ __forceinline__ v16h frag_a(const _Float16* A, int lda, int m0, int k0, int lane) {
  int row = m0 + (lane & 15);
  const _Float16* p = A + (size_t)row * lda + k0 + ((lane >> 4) << 3);
  H16u u; u.h[0] = *(const half8*)p; u.h[1] = *(const half8*)(p + 16);
  return u.v;
}
// B (32x16 f16) from transposed weights Bt[n*ldb + k]:
// lane<16 -> col n=lane, K = k0..k0+7 and k0+16..k0+23; lane>=16 -> K = k0+8..k0+15 and k0+24..k0+31.
__device__ __forceinline__ v16h frag_b(const _Float16* Bt, int ldb, int n0, int k0, int lane) {
  int n = n0 + (lane & 15);
  const _Float16* p = Bt + (size_t)n * ldb + k0 + ((lane >> 4) << 3);
  H16u u; u.h[0] = *(const half8*)p; u.h[1] = *(const half8*)(p + 16);
  return u.v;
}

// ---------------- elementwise helpers ----------------
__global__ void k_copy_f32(float* __restrict__ d, const float* __restrict__ s, int n) {
  int i = blockIdx.x * blockDim.x + threadIdx.x;
  if (i < n) d[i] = s[i];
}
__global__ void k_f32_to_f16(_Float16* __restrict__ d, const float* __restrict__ s, int n) {
  int i = blockIdx.x * blockDim.x + threadIdx.x;
  if (i < n) d[i] = (_Float16)s[i];
}
// W is KxN row-major; Wt is Npad x K row-major f16, zero pad rows >= N.
__global__ void k_transpose_f16(const float* __restrict__ W, _Float16* __restrict__ Wt,
                                int K, int N, int Npad) {
  int i = blockIdx.x * blockDim.x + threadIdx.x;
  if (i >= Npad * K) return;
  int n = i / K, k = i - n * K;
  Wt[i] = (n < N) ? (_Float16)W[(size_t)k * N + n] : (_Float16)0.0f;
}

// ---------------- GIN scatter: agg[dst] += h[src] ----------------
__global__ void k_scatter_add(float* __restrict__ agg, const float* __restrict__ h,
                              const int* __restrict__ src, const int* __restrict__ dst, int F) {
  int idx = blockIdx.x * blockDim.x + threadIdx.x;
  if (idx >= E_ * F) return;
  int e = idx / F, f = idx - e * F;
  atomicAdd(&agg[(size_t)dst[e] * F + f], h[(size_t)src[e] * F + f]);
}

// ---------------- generic WMMA GEMM: C = act(A(f16,MxK) @ Bt^T + bias) ----------------
__global__ __launch_bounds__(256) void k_gemm_wmma(
    const _Float16* __restrict__ A, const _Float16* __restrict__ Bt,
    const float* __restrict__ bias, float* __restrict__ Cf, _Float16* __restrict__ Ch,
    int M, int K, int N, int ntiles, int relu) {
  int lane = threadIdx.x & 31;
  int gw = blockIdx.x * (blockDim.x >> 5) + (threadIdx.x >> 5);
  int mtiles = M >> 4;
  if (gw >= mtiles * ntiles) return;           // wave-uniform exit (EXEC stays full)
  int mt = gw / ntiles, nt = gw - mt * ntiles;
  v8f acc = {};
  for (int k0 = 0; k0 < K; k0 += 32) {
    v16h a = frag_a(A, K, mt << 4, k0, lane);
    v16h b = frag_b(Bt, K, nt << 4, k0, lane);
    acc = __builtin_amdgcn_wmma_f32_16x16x32_f16(false, a, false, b, (short)0, acc, false, false);
  }
  int nc = (nt << 4) + (lane & 15);
  int hi = lane >> 4;
  if (nc < N) {
    float bv = bias ? bias[nc] : 0.0f;
#pragma unroll
    for (int r = 0; r < 8; r++) {
      int m = (mt << 4) + hi * 8 + r;
      float v = acc[r] + bv;
      if (relu) v = fmaxf(v, 0.0f);
      if (Cf) Cf[(size_t)m * N + nc] = v;
      if (Ch) Ch[(size_t)m * N + nc] = (_Float16)v;
    }
  }
}

// ---------------- BatchNorm (training stats) over 2048 nodes, per-feature block ----------------
__global__ __launch_bounds__(256) void k_bn(float* __restrict__ h, const float* __restrict__ gamma,
                                            const float* __restrict__ beta, int relu) {
  int f = blockIdx.x, tid = threadIdx.x;
  __shared__ float rs[256], rq[256];
  float s = 0.f, q = 0.f;
  for (int i = tid; i < NN_; i += 256) { float v = h[(size_t)i * 128 + f]; s += v; q += v * v; }
  rs[tid] = s; rq[tid] = q; __syncthreads();
  for (int o = 128; o > 0; o >>= 1) {
    if (tid < o) { rs[tid] += rs[tid + o]; rq[tid] += rq[tid + o]; }
    __syncthreads();
  }
  float mu = rs[0] * (1.0f / NN_);
  float var = rq[0] * (1.0f / NN_) - mu * mu;
  float sc = rsqrtf(var + 1e-5f) * gamma[f];
  float bt = beta[f];
  for (int i = tid; i < NN_; i += 256) {
    float v = (h[(size_t)i * 128 + f] - mu) * sc + bt;
    if (relu) v = fmaxf(v, 0.0f);
    h[(size_t)i * 128 + f] = v;
  }
}

// ---------------- global mean pool -> z (f32 + f16) ----------------
__global__ void k_pool(const float* __restrict__ h, float* __restrict__ z, _Float16* __restrict__ zh) {
  int i = blockIdx.x * blockDim.x + threadIdx.x;
  if (i >= BATCH * LATENT) return;
  int g = i >> 7, f = i & 127;
  float s = 0.f;
  for (int n = 0; n < NBG; n++) s += h[((size_t)g * NBG + n) * 128 + f];
  float v = s * (1.0f / NBG);
  z[i] = v; zh[i] = (_Float16)v;
}

// ---------------- modulator MLP: mods[5][B][128] ----------------
__global__ __launch_bounds__(128) void k_modulator(
    const float* __restrict__ z,
    const float* W0, const float* W1, const float* W2, const float* W3, const float* W4,
    const float* b0, const float* b1, const float* b2, const float* b3, const float* b4,
    float* __restrict__ mods) {
  int g = blockIdx.x, f = threadIdx.x;
  __shared__ float zs[128], ms[128];
  zs[f] = z[g * 128 + f]; __syncthreads();
  const float* Ws[5] = {W0, W1, W2, W3, W4};
  const float* bs[5] = {b0, b1, b2, b3, b4};
  float a = bs[0][f];
  for (int k = 0; k < 128; k++) a += zs[k] * Ws[0][k * 128 + f];
  float m = fmaxf(a, 0.0f);
  mods[(0 * BATCH + g) * 128 + f] = m;
  __syncthreads(); ms[f] = m; __syncthreads();
  for (int l = 1; l < 5; l++) {               // 256 -> 128 with [m, z] concat
    a = bs[l][f];
    for (int k = 0; k < 128; k++) a += ms[k] * Ws[l][k * 128 + f];
    for (int k = 0; k < 128; k++) a += zs[k] * Ws[l][(128 + k) * 128 + f];
    m = fmaxf(a, 0.0f);
    mods[(l * BATCH + g) * 128 + f] = m;
    __syncthreads(); ms[f] = m; __syncthreads();
  }
}

// ---------------- fused modulated-SIREN kernel (the hot loop) ----------------
// Block = 128 threads / 4 waves, owns a 64-row slab of grid points.
// Per layer, the 128x128 f16 weight tile is DMA'd into LDS by the Tensor Data
// Mover (tensor_load_to_lds) with pad_enable inserting 16B per 256B row so the
// staged tile lands at a bank-conflict-free 272B stride.
#define HSTR 136   // padded row stride in f16 (272B)
#define ROWS 64

__device__ __forceinline__ void stage_weights_tdm(const _Float16* Wt, _Float16* Wst,
                                                  int wave, int tid) {
#if __has_builtin(__builtin_amdgcn_tensor_load_to_lds) && __has_builtin(__builtin_amdgcn_s_wait_tensorcnt)
  if (wave == 0) {
    unsigned lds = (unsigned)(size_t)(&Wst[0]);
    unsigned long long ga = (unsigned long long)(size_t)Wt;
    u32x4 g0;
    g0.x = 1u;                                           // count=1 (valid user D#)
    g0.y = lds;                                          // lds_addr
    g0.z = (unsigned)(ga & 0xFFFFFFFFu);                 // global_addr[31:0]
    g0.w = (unsigned)((ga >> 32) & 0x01FFFFFFu) | (2u << 30); // addr[56:32] | type=2
    i32x8 g1;
    g1[0] = (1 << 16)      // data_size = 2 bytes
          | (1 << 20)      // pad_enable
          | (5 << 22)      // pad_interval: every 64 DWORDs (256B)
          | (3 << 25);     // pad_amount: 4 DWORDs (16B)
    g1[1] = (128 << 16);   // tensor_dim0 = 128 (low 16)
    g1[2] = (128 << 16);   // tensor_dim0 hi=0 | tensor_dim1 = 128 (low 16)
    g1[3] = (128 << 16);   // tensor_dim1 hi=0 | tile_dim0 = 128
    g1[4] = 128;           // tile_dim1 = 128, tile_dim2 = 0 (2D)
    g1[5] = 128;           // tensor_dim0_stride = 128 (low 32)
    g1[6] = 0;             // stride hi | tensor_dim1_stride lo (unused, 2D)
    g1[7] = 0;
    i32x4 gz = {0, 0, 0, 0};
    i32x8 gz8 = {0, 0, 0, 0, 0, 0, 0, 0};
    // 6-arg toolchain signature: (g0, g1, g2, g3, g4, cpol)
    __builtin_amdgcn_tensor_load_to_lds(g0, g1, gz, gz, gz8, 0);
    __builtin_amdgcn_s_wait_tensorcnt(0);                // TENSORcnt == 0
  }
#else
  // fallback: cooperative vector copy with manual 272B-stride padding
  for (int i = tid; i < 128 * 16; i += 128) {
    int row = i >> 4, ch = i & 15;
    *(half8*)&Wst[row * HSTR + ch * 8] = *(const half8*)(Wt + (size_t)row * 128 + ch * 8);
  }
#endif
  __syncthreads();
}

__device__ __forceinline__ v16h frag_b_lds(const _Float16* Bt, int n0, int k0, int lane) {
  const _Float16* p = Bt + (size_t)(n0 + (lane & 15)) * HSTR + k0 + ((lane >> 4) << 3);
  H16u u; u.h[0] = *(const half8*)p; u.h[1] = *(const half8*)(p + 16);
  return u.v;
}

__global__ __launch_bounds__(128) void k_siren(
    const float* __restrict__ W0, const float* __restrict__ b0,
    const _Float16* Wt1, const float* b1, const _Float16* Wt2, const float* b2,
    const _Float16* Wt3, const float* b3, const _Float16* Wt4, const float* b4,
    const float* __restrict__ Wout, const float* __restrict__ bout,
    const float* __restrict__ mods, float* __restrict__ Craw) {
  __shared__ _Float16 hh[ROWS * HSTR];     // 17,408 B activation slab
  __shared__ _Float16 Wst[128 * HSTR];     // 34,816 B TDM-staged weight tile
  int g = blockIdx.y, chunk = blockIdx.x, tid = threadIdx.x;
  int rowbase = chunk * ROWS;              // grid-point base for this block
  // ---- layer 0: grid(2) -> 128, w0 = 30 ----
  for (int idx = tid; idx < ROWS * 128; idx += 128) {
    int r = idx >> 7, c = idx & 127;
    int p = rowbase + r;
    float xg = ((float)(p >> 7) + 0.5f) * (1.0f / NBG);
    float yg = ((float)(p & 127) + 0.5f) * (1.0f / NBG);
    float v = sinf(30.0f * (xg * W0[c] + yg * W0[128 + c] + b0[c]));
    v *= mods[(0 * BATCH + g) * 128 + c];
    hh[r * HSTR + c] = (_Float16)v;
  }
  __syncthreads();
  const _Float16* Wts[4] = {Wt1, Wt2, Wt3, Wt4};
  const float* bs[4] = {b1, b2, b3, b4};
  int wave = tid >> 5, lane = tid & 31;
  int arow = wave * 16 + (lane & 15);
  int kb = (lane >> 4) << 3;
#pragma unroll
  for (int l = 0; l < 4; l++) {
    stage_weights_tdm(Wts[l], Wst, wave, tid);   // TDM -> LDS, + barrier
    if (l < 3) __builtin_prefetch(Wts[l + 1] + (size_t)lane * 128, 0, 3);
    // wave's 16x128 activation slab -> 4 A fragments (entirely in VGPRs)
    v16h afr[4];
#pragma unroll
    for (int kt = 0; kt < 4; kt++) {
      const _Float16* p = &hh[arow * HSTR + kt * 32 + kb];
      H16u u; u.h[0] = *(const half8*)p; u.h[1] = *(const half8*)(p + 16);
      afr[kt] = u.v;
    }
    const float* bb = bs[l];
#pragma unroll
    for (int nt = 0; nt < 8; nt++) {
      v8f acc = {};
#pragma unroll
      for (int kt = 0; kt < 4; kt++) {
        v16h b = frag_b_lds(Wst, nt << 4, kt * 32, lane);
        acc = __builtin_amdgcn_wmma_f32_16x16x32_f16(false, afr[kt], false, b,
                                                     (short)0, acc, false, false);
      }
      int n = (nt << 4) + (lane & 15);
      float bv = bb[n];
      float mv = mods[((l + 1) * BATCH + g) * 128 + n];
      int mbase = wave * 16 + (lane >> 4) * 8;
#pragma unroll
      for (int r = 0; r < 8; r++) {
        float v = sinf(acc[r] + bv) * mv;      // w0 = 1 for hidden layers
        hh[(mbase + r) * HSTR + n] = (_Float16)v;
      }
    }
    __syncthreads();   // all waves done with Wst before next layer's TDM
  }
  // ---- output head: 128 -> 1, sigmoid ----
  if (tid < ROWS) {
    float s = bout[0];
    for (int k = 0; k < 128; k++) s += (float)hh[tid * HSTR + k] * Wout[k];
    Craw[(size_t)g * GRIDPTS + rowbase + tid] = 1.0f / (1.0f + expf(-s));
  }
}

// ---------------- C = triu(C,1) + triu(C,1)^T ----------------
__global__ void k_symmetrize(const float* __restrict__ Craw, float* __restrict__ Csym) {
  int idx = blockIdx.x * blockDim.x + threadIdx.x;
  if (idx >= BATCH * GRIDPTS) return;
  int g = idx >> 14, r = (idx >> 7) & 127, c = idx & 127;
  float v = 0.0f;
  if (c > r) v = Craw[(size_t)g * GRIDPTS + r * 128 + c];
  else if (r > c) v = Craw[(size_t)g * GRIDPTS + c * 128 + r];
  Csym[idx] = v;
}

// ---------------- deterministic on-device THETA (column-normalized normals) ----------------
__device__ __forceinline__ unsigned int hashu(unsigned int x) {
  x ^= x >> 16; x *= 0x7feb352du; x ^= x >> 15; x *= 0x846ca68bu; x ^= x >> 16; return x;
}
__device__ __forceinline__ float dnormal(unsigned int s) {
  unsigned int a = hashu(s * 2654435761u + 0x9E3779B9u);
  unsigned int b = hashu(a ^ 0x85EBCA6Bu);
  float u1 = ((float)a + 1.0f) * 2.3283064e-10f;
  float u2 = (float)b * 2.3283064e-10f;
  return sqrtf(-2.0f * logf(u1)) * cosf(6.28318530718f * u2);
}
__global__ void k_theta(float* __restrict__ theta) {
  int p = threadIdx.x;
  if (p >= NPROJ_) return;
  float ss = 0.0f;
  for (int i = 0; i < 128; i++) { float v = dnormal(i * NPROJ_ + p); ss += v * v; }
  float inv = rsqrtf(ss + 1e-12f);
  for (int i = 0; i < 128; i++) theta[i * NPROJ_ + p] = dnormal(i * NPROJ_ + p) * inv;
}

// ---------------- projections: P[g,i,p] = sum_j src[g,i,j] * theta[j,p] ----------------
__global__ void k_project(const float* __restrict__ src, const float* __restrict__ theta,
                          float* __restrict__ P) {
  int idx = blockIdx.x * blockDim.x + threadIdx.x;
  if (idx >= BATCH * 128 * NPROJ_) return;
  int p = idx % NPROJ_;
  int gi = idx / NPROJ_;               // g*128+i
  const float* row = src + (size_t)gi * 128;
  float s = 0.0f;
  for (int j = 0; j < 128; j++) s += row[j] * theta[j * NPROJ_ + p];
  P[idx] = s;
}

// ---------------- bitonic sort of 128-element columns (in-place) ----------------
__global__ __launch_bounds__(128) void k_sort128(float* __restrict__ P, float* __restrict__ Pin) {
  __shared__ float s[128];
  int b = blockIdx.x, tid = threadIdx.x;
  float* M = (b < BATCH * NPROJ_) ? P : Pin;
  int col = b % (BATCH * NPROJ_);
  int g = col / NPROJ_, p = col % NPROJ_;
  size_t base = ((size_t)g * 128) * NPROJ_ + p;
  s[tid] = M[base + (size_t)tid * NPROJ_];
  __syncthreads();
  for (int k = 2; k <= 128; k <<= 1)
    for (int j = k >> 1; j > 0; j >>= 1) {
      int ixj = tid ^ j;
      if (ixj > tid) {
        bool up = ((tid & k) == 0);
        float a = s[tid], c = s[ixj];
        if ((a > c) == up) { s[tid] = c; s[ixj] = a; }
      }
      __syncthreads();
    }
  M[base + (size_t)tid * NPROJ_] = s[tid];
}

// ---------------- per-graph sliced-W2 loss, then mean ----------------
__global__ __launch_bounds__(256) void k_loss(const float* __restrict__ P,
                                              const float* __restrict__ Pin,
                                              float* __restrict__ lossP) {
  int g = blockIdx.x, tid = threadIdx.x;
  __shared__ float rs[256];
  float s = 0.0f;
  for (int i = tid; i < 128 * NPROJ_; i += 256) {
    float d = P[(size_t)g * 128 * NPROJ_ + i] - Pin[(size_t)g * 128 * NPROJ_ + i];
    s += d * d;
  }
  rs[tid] = s; __syncthreads();
  for (int o = 128; o > 0; o >>= 1) { if (tid < o) rs[tid] += rs[tid + o]; __syncthreads(); }
  if (tid == 0) lossP[g] = sqrtf(rs[0] * (1.0f / (128.0f * NPROJ_)));
}
__global__ void k_finalize(const float* __restrict__ lossP, float* __restrict__ out) {
  if (threadIdx.x == 0) {
    float s = 0.0f;
    for (int g = 0; g < BATCH; g++) s += lossP[g];
    out[0] = s * (1.0f / BATCH);
  }
}

// ---------------- host orchestration ----------------
static inline void* ws_take(char*& w, size_t bytes) {
  void* p = (void*)w;
  w += (bytes + 255) & ~(size_t)255;
  return p;
}

extern "C" void kernel_launch(void* const* d_in, const int* in_sizes, int n_in,
                              void* d_out, int out_size, void* d_ws, size_t ws_size,
                              hipStream_t stream) {
  (void)in_sizes; (void)n_in; (void)out_size; (void)ws_size;
  // inputs: x, edge_index, batch, C_input, M, then params flattened as a JAX
  // pytree (dict keys sorted): bn, coords, gin, mod, siren.
  const float* x   = (const float*)d_in[0];
  const int*   ei  = (const int*)d_in[1];
  const float* Cin = (const float*)d_in[3];
  const float *bnBeta[3], *bnGamma[3];
  for (int l = 0; l < 3; l++) { bnBeta[l] = (const float*)d_in[5 + 2 * l];
                                bnGamma[l] = (const float*)d_in[6 + 2 * l]; }
  const float* coordsW = (const float*)d_in[11];
  const float* coordsB = (const float*)d_in[12];
  const float *gW1[3], *gW2[3], *gb1[3], *gb2[3];
  for (int l = 0; l < 3; l++) {
    gW1[l] = (const float*)d_in[13 + 4 * l]; gW2[l] = (const float*)d_in[14 + 4 * l];
    gb1[l] = (const float*)d_in[15 + 4 * l]; gb2[l] = (const float*)d_in[16 + 4 * l];
  }
  const float *modW[5], *modB[5];
  for (int j = 0; j < 5; j++) { modW[j] = (const float*)d_in[25 + j];
                                modB[j] = (const float*)d_in[30 + j]; }
  const float* sirW[5]; for (int j = 0; j < 5; j++) sirW[j] = (const float*)d_in[35 + j];
  const float* sirWout = (const float*)d_in[40];
  const float* sirB[5]; for (int j = 0; j < 5; j++) sirB[j] = (const float*)d_in[41 + j];
  const float* sirBout = (const float*)d_in[46];
  float* out = (float*)d_out;

  // workspace carve-up
  char* w = (char*)d_ws;
  _Float16* ginW1t[3];
  ginW1t[0] = (_Float16*)ws_take(w, (size_t)128 * 64 * 2);
  ginW1t[1] = (_Float16*)ws_take(w, (size_t)128 * 128 * 2);
  ginW1t[2] = (_Float16*)ws_take(w, (size_t)128 * 128 * 2);
  _Float16* ginW2t[3];
  for (int l = 0; l < 3; l++) ginW2t[l] = (_Float16*)ws_take(w, (size_t)128 * 128 * 2);
  _Float16* sirWt[4];
  for (int j = 0; j < 4; j++) sirWt[j] = (_Float16*)ws_take(w, (size_t)128 * 128 * 2);
  _Float16* coordsWt = (_Float16*)ws_take(w, (size_t)NCP_ * 128 * 2);
  _Float16* zh   = (_Float16*)ws_take(w, (size_t)BATCH * 128 * 2);
  float*    agg  = (float*)ws_take(w, (size_t)NN_ * 128 * 4);
  _Float16* aggH = (_Float16*)ws_take(w, (size_t)NN_ * 128 * 2);
  _Float16* t1H  = (_Float16*)ws_take(w, (size_t)NN_ * 128 * 2);
  float*    hbuf = (float*)ws_take(w, (size_t)NN_ * 128 * 4);
  float*    zf   = (float*)ws_take(w, (size_t)BATCH * 128 * 4);
  float*    mods = (float*)ws_take(w, (size_t)5 * BATCH * 128 * 4);
  float*    theta= (float*)ws_take(w, (size_t)128 * NPROJ_ * 4);
  float*    Craw = (float*)ws_take(w, (size_t)BATCH * GRIDPTS * 4);
  float*    Csym = (float*)ws_take(w, (size_t)BATCH * GRIDPTS * 4);
  float*    P    = (float*)ws_take(w, (size_t)BATCH * 128 * NPROJ_ * 4);
  float*    Pin  = (float*)ws_take(w, (size_t)BATCH * 128 * NPROJ_ * 4);
  float*    lossP= (float*)ws_take(w, 64);

  const int T = 256;
  auto blocks = [](int n, int t) { return (n + t - 1) / t; };

  // ---- weight repack f32 -> transposed f16 ----
  k_transpose_f16<<<blocks(128 * 64, T), T, 0, stream>>>(gW1[0], ginW1t[0], 64, 128, 128);
  k_transpose_f16<<<blocks(128 * 128, T), T, 0, stream>>>(gW1[1], ginW1t[1], 128, 128, 128);
  k_transpose_f16<<<blocks(128 * 128, T), T, 0, stream>>>(gW1[2], ginW1t[2], 128, 128, 128);
  for (int l = 0; l < 3; l++)
    k_transpose_f16<<<blocks(128 * 128, T), T, 0, stream>>>(gW2[l], ginW2t[l], 128, 128, 128);
  for (int j = 0; j < 4; j++)
    k_transpose_f16<<<blocks(128 * 128, T), T, 0, stream>>>(sirW[j + 1], sirWt[j], 128, 128, 128);
  k_transpose_f16<<<blocks(NCP_ * 128, T), T, 0, stream>>>(coordsW, coordsWt, 128, NC_, NCP_);
  k_theta<<<1, 64, 0, stream>>>(theta);

  // ---- GIN encoder: 3 layers ----
  for (int l = 0; l < 3; l++) {
    const float* hin = (l == 0) ? x : hbuf;
    int K = (l == 0) ? 64 : 128;
    k_copy_f32<<<blocks(NN_ * K, T), T, 0, stream>>>(agg, hin, NN_ * K);
    k_scatter_add<<<blocks(E_ * K, T), T, 0, stream>>>(agg, hin, ei, ei + E_, K);
    k_f32_to_f16<<<blocks(NN_ * K, T), T, 0, stream>>>(aggH, agg, NN_ * K);
    {   // t1 = relu(agg @ W1 + b1)  [WMMA]
      int tiles = (NN_ / 16) * (128 / 16);
      k_gemm_wmma<<<blocks(tiles, 8), T, 0, stream>>>(aggH, ginW1t[l], gb1[l],
                                                      nullptr, t1H, NN_, K, 128, 128 / 16, 1);
    }
    {   // h = t1 @ W2 + b2  [WMMA]
      int tiles = (NN_ / 16) * (128 / 16);
      k_gemm_wmma<<<blocks(tiles, 8), T, 0, stream>>>(t1H, ginW2t[l], gb2[l],
                                                      hbuf, nullptr, NN_, 128, 128, 128 / 16, 0);
    }
    k_bn<<<128, T, 0, stream>>>(hbuf, bnGamma[l], bnBeta[l], (l < 2) ? 1 : 0);
  }

  // ---- pool + modulator ----
  k_pool<<<blocks(BATCH * 128, T), T, 0, stream>>>(hbuf, zf, zh);
  k_modulator<<<BATCH, 128, 0, stream>>>(zf, modW[0], modW[1], modW[2], modW[3], modW[4],
                                         modB[0], modB[1], modB[2], modB[3], modB[4], mods);

  // ---- fused modulated SIREN over the 128x128 grid, per graph ----
  {
    dim3 grid(GRIDPTS / ROWS, BATCH);
    k_siren<<<grid, 128, 0, stream>>>(sirW[0], sirB[0],
                                      sirWt[0], sirB[1], sirWt[1], sirB[2],
                                      sirWt[2], sirB[3], sirWt[3], sirB[4],
                                      sirWout, sirBout, mods, Craw);
  }

  // ---- symmetrize, project, sort, loss ----
  k_symmetrize<<<blocks(BATCH * GRIDPTS, T), T, 0, stream>>>(Craw, Csym);
  k_project<<<blocks(BATCH * 128 * NPROJ_, T), T, 0, stream>>>(Csym, theta, P);
  k_project<<<blocks(BATCH * 128 * NPROJ_, T), T, 0, stream>>>(Cin, theta, Pin);
  k_sort128<<<2 * BATCH * NPROJ_, 128, 0, stream>>>(P, Pin);
  k_loss<<<BATCH, T, 0, stream>>>(P, Pin, lossP);
  k_finalize<<<1, 32, 0, stream>>>(lossP, out);

  // ---- coords head: (16 x 128) @ (128 x 6573) + b  [WMMA], into d_out+1 ----
  {
    int ntiles = NCP_ / 16;                  // 411
    int tiles = (16 / 16) * ntiles;
    k_gemm_wmma<<<blocks(tiles, 8), T, 0, stream>>>(zh, coordsWt, coordsB,
                                                    out + 1, nullptr, 16, 128, NC_, ntiles, 0);
  }
}
